// ScaledDotProductAttention_90804198572610
// MI455X (gfx1250) — matrix operand order repaired and checked by reading, not compiled, not run on hardware
//
#include <hip/hip_runtime.h>

// ---------------------------------------------------------------------------
// Scaled dot-product attention for gfx1250 (MI455X), wave32 + WMMA bf16.
// B=16, L=2048, D=64.  d_out = [output (B*L*D) | attention (B*L*L)] fp32.
//
// v_wmma_f32_16x16x32_bf16 for QK^T and P*V (bf16 in, fp32 accum).
// Two-pass safe softmax in the log2 domain: scale*log2(e) folded into Q at
// bf16 conversion, so softmax needs only v_exp_f32/v_log_f32 + adds.
// Double-buffered LDS staging: one barrier per key tile, global loads of
// tile kt+1 overlap WMMA on tile kt; global_prefetch_b8 at kt+2.
// Attention matrix (268 MB) written with nontemporal stores.
// ---------------------------------------------------------------------------

typedef __attribute__((ext_vector_type(16))) __bf16          v16bf;
typedef __attribute__((ext_vector_type(8)))  float           v8f;
typedef __attribute__((ext_vector_type(8)))  unsigned short  us8;   // 16 bytes
typedef __attribute__((ext_vector_type(4)))  unsigned short  us4;   // 8 bytes

union Frag {
    v16bf v;
    us8   h[2];
};

__device__ __forceinline__ unsigned short f2bf(float f) {
    // round-to-nearest-even fp32 -> bf16
    unsigned int u = __float_as_uint(f);
    u += 0x7FFFu + ((u >> 16) & 1u);
    return (unsigned short)(u >> 16);
}

__device__ __forceinline__ float fast_exp2(float x) {
#if __has_builtin(__builtin_amdgcn_exp2f)
    return __builtin_amdgcn_exp2f(x);      // v_exp_f32
#else
    return exp2f(x);
#endif
}

__device__ __forceinline__ float fast_log2(float x) {
#if __has_builtin(__builtin_amdgcn_logf)
    return __builtin_amdgcn_logf(x);       // v_log_f32
#else
    return log2f(x);
#endif
}

__device__ __forceinline__ v8f wmma_bf16(const Frag& a, const Frag& b, v8f c) {
    // (neg_a, A, neg_b, B, c_mod, C, reuse_a, reuse_b)
    return __builtin_amdgcn_wmma_f32_16x16x32_bf16(false, a.v, false, b.v,
                                                   (short)0, c, false, false);
}

__global__ __launch_bounds__(256)
void ScaledDotProductAttention_90804198572610_kernel(
    const float* __restrict__ Qg, const float* __restrict__ Kg,
    const float* __restrict__ Vg, float* __restrict__ Og,
    float* __restrict__ Ag)
{
    constexpr int L  = 2048;
    constexpr int D  = 64;
    constexpr int BQ = 128;          // queries per block (8 waves x 16)
    constexpr int TK = 32;           // keys per iteration (one bf16 WMMA K-dim)
    constexpr int NKT = L / TK;      // 64 iterations
    // fold 1/sqrt(64) and log2(e) into Q so scores live in the log2 domain
    constexpr float qscale = 0.125f * 1.4426950408889634f;

    __shared__ unsigned short Qs[BQ * D];        // 16 KB bf16 Q tile
    __shared__ unsigned short Ks[2][TK * D];     //  8 KB ping-pong K tiles
    __shared__ unsigned short Vt[2][D * TK];     //  8 KB ping-pong V^T tiles
    __shared__ unsigned short Ps[8 * 16 * TK];   //  8 KB per-wave P staging

    const int tid  = threadIdx.x;
    const int lane = tid & 31;
    const int wv   = tid >> 5;       // wave id 0..7
    const int hf   = lane >> 4;      // half-wave 0/1
    const int ln   = lane & 15;      // lane within half

    const int b  = blockIdx.x / (L / BQ);
    const int qb = blockIdx.x % (L / BQ);

    const float* Qbase = Qg + ((size_t)b * L + (size_t)qb * BQ) * D;
    const float* Kbase = Kg + (size_t)b * L * D;
    const float* Vbase = Vg + (size_t)b * L * D;
    float*       attn  = Ag + (size_t)b * L * L;

    // ---- staging helpers --------------------------------------------------
    auto stageK = [&](int kt, int buf) {
        const float4* k4 = (const float4*)(Kbase + (size_t)kt * TK * D);
        unsigned short* dst = Ks[buf];
        #pragma unroll
        for (int j = 0; j < (TK * D / 4) / 256; ++j) {
            const int i = tid + j * 256;
            float4 f = k4[i];
            us4 t = { f2bf(f.x), f2bf(f.y), f2bf(f.z), f2bf(f.w) };
            *(us4*)&dst[i * 4] = t;
        }
    };
    auto stageV = [&](int kt, int buf) {
        const float4* v4 = (const float4*)(Vbase + (size_t)kt * TK * D);
        unsigned short* dst = Vt[buf];
        #pragma unroll
        for (int j = 0; j < (TK * D / 4) / 256; ++j) {
            const int i = tid + j * 256;
            float4 g = v4[i];
            const int r = i >> 4;           // key row (16 float4 per row)
            const int d = (i & 15) * 4;     // d offset
            dst[(d + 0) * TK + r] = f2bf(g.x);   // transpose -> [d][k]
            dst[(d + 1) * TK + r] = f2bf(g.y);
            dst[(d + 2) * TK + r] = f2bf(g.z);
            dst[(d + 3) * TK + r] = f2bf(g.w);
        }
    };

    // ---- stage Q tile (BQ x D), scaled, fp32 -> bf16 LDS ------------------
    {
        const float4* q4 = (const float4*)Qbase;
        #pragma unroll
        for (int j = 0; j < (BQ * D / 4) / 256; ++j) {
            const int i = tid + j * 256;
            float4 f = q4[i];
            us4 t = { f2bf(f.x * qscale), f2bf(f.y * qscale),
                      f2bf(f.z * qscale), f2bf(f.w * qscale) };
            *(us4*)&Qs[i * 4] = t;
        }
    }
    stageK(0, 0);                     // prologue for pass-1 double buffer
    __syncthreads();

    // ---- persistent Q A-fragments (16x32 bf16 layout, M = ln) -------------
    Frag aQ0, aQ1;
    {
        const int row = (wv * 16 + ln) * D;
        const int off = hf * 8;
        aQ0.h[0] = *(const us8*)&Qs[row + off];           // K = off..off+7
        aQ0.h[1] = *(const us8*)&Qs[row + 16 + off];      // K = 16+off..
        aQ1.h[0] = *(const us8*)&Qs[row + 32 + off];
        aQ1.h[1] = *(const us8*)&Qs[row + 32 + 16 + off];
    }

    float mrow[8], lrow[8], crow[8];
    #pragma unroll
    for (int v = 0; v < 8; ++v) { mrow[v] = -3.0e38f; lrow[v] = 0.0f; }

    // ======================= PASS 1: row max / sum2^ =======================
    for (int kt = 0; kt < NKT; ++kt) {
        __syncthreads();              // buf[kt&1] ready; buf[(kt+1)&1] free
        if (kt + 1 < NKT) stageK(kt + 1, (kt + 1) & 1);
        if (kt + 2 < NKT)
            __builtin_prefetch(Kbase + (size_t)(kt + 2) * TK * D + tid * 8, 0, 1);

        const unsigned short* Kb = Ks[kt & 1];
        #pragma unroll
        for (int st = 0; st < 2; ++st) {
            Frag b0, b1;                               // B = K^T (32d x 16k)
            const int krow = (st * 16 + ln) * D;       // N = key = ln
            b0.h[0] = *(const us8*)&Kb[krow + hf * 8];
            b0.h[1] = *(const us8*)&Kb[krow + 16 + hf * 8];
            b1.h[0] = *(const us8*)&Kb[krow + 32 + hf * 8];
            b1.h[1] = *(const us8*)&Kb[krow + 32 + 16 + hf * 8];

            v8f acc = {};
            acc = wmma_bf16(aQ0, b0, acc);
            acc = wmma_bf16(aQ1, b1, acc);

            #pragma unroll
            for (int v = 0; v < 8; ++v) {
                const float s  = acc[v];               // log2-domain score
                const float mo = mrow[v];
                const float mn = fmaxf(mo, s);
                const float e  = fast_exp2(fminf(mo, s) - mn);  // single exp
                lrow[v] = (s > mo) ? (lrow[v] * e + 1.0f) : (lrow[v] + e);
                mrow[v] = mn;
            }
        }
    }

    // ---- reduce stats across the 16 lanes of each half --------------------
    #pragma unroll
    for (int v = 0; v < 8; ++v) {
        float m = mrow[v], l = lrow[v];
        #pragma unroll
        for (int off = 8; off >= 1; off >>= 1) {
            const float m2 = __shfl_xor(m, off, 16);
            const float l2 = __shfl_xor(l, off, 16);
            const float mn = fmaxf(m, m2);
            const float e  = fast_exp2(fminf(m, m2) - mn);
            l = (m >= m2) ? (l + l2 * e) : (l * e + l2);
            m = mn;
        }
        crow[v] = m + fast_log2(l);    // p = 2^(s - crow), already normalized
    }

    // ======================= PASS 2: probs + O = P*V =======================
    v8f oacc[4] = {};

    stageK(0, 0);
    stageV(0, 0);                     // prologue for pass-2 double buffer
    for (int kt = 0; kt < NKT; ++kt) {
        __syncthreads();
        if (kt + 1 < NKT) { stageK(kt + 1, (kt + 1) & 1); stageV(kt + 1, (kt + 1) & 1); }
        if (kt + 2 < NKT) {
            __builtin_prefetch(Kbase + (size_t)(kt + 2) * TK * D + tid * 8, 0, 1);
            __builtin_prefetch(Vbase + (size_t)(kt + 2) * TK * D + tid * 8, 0, 1);
        }

        const unsigned short* Kb = Ks[kt & 1];
        const unsigned short* Vb = Vt[kt & 1];

        #pragma unroll
        for (int st = 0; st < 2; ++st) {
            Frag b0, b1;
            const int krow = (st * 16 + ln) * D;
            b0.h[0] = *(const us8*)&Kb[krow + hf * 8];
            b0.h[1] = *(const us8*)&Kb[krow + 16 + hf * 8];
            b1.h[0] = *(const us8*)&Kb[krow + 32 + hf * 8];
            b1.h[1] = *(const us8*)&Kb[krow + 32 + 16 + hf * 8];

            v8f acc = {};
            acc = wmma_bf16(aQ0, b0, acc);
            acc = wmma_bf16(aQ1, b1, acc);

            #pragma unroll
            for (int v = 0; v < 8; ++v) {
                const float p = fast_exp2(acc[v] - crow[v]);   // normalized prob
                const int   m = v + 8 * hf;                    // row in C tile
                const size_t qrow = (size_t)qb * BQ + wv * 16 + m;
                __builtin_nontemporal_store(
                    p, &attn[qrow * L + (size_t)kt * TK + st * 16 + ln]);
                Ps[(wv * 16 + m) * TK + st * 16 + ln] = f2bf(p);
            }
        }

        // within-wave LDS RAW: make P staging visible before fragment reads
        asm volatile("s_wait_dscnt 0" ::: "memory");

        Frag aP;                       // P as A (16x32 bf16), M = ln
        {
            const int base = (wv * 16 + ln) * TK;
            const int off  = hf * 8;
            aP.h[0] = *(const us8*)&Ps[base + off];
            aP.h[1] = *(const us8*)&Ps[base + 16 + off];
        }

        #pragma unroll
        for (int c = 0; c < 4; ++c) {
            Frag bv;                   // V as B (32k x 16d), N = d = c*16+ln
            const int col = (c * 16 + ln) * TK;
            bv.h[0] = *(const us8*)&Vb[col + hf * 8];
            bv.h[1] = *(const us8*)&Vb[col + 16 + hf * 8];
            oacc[c] = wmma_bf16(aP, bv, oacc[c]);
        }
    }

    // ---- store O (P was normalized, so O is final) ------------------------
    #pragma unroll
    for (int c = 0; c < 4; ++c) {
        #pragma unroll
        for (int v = 0; v < 8; ++v) {
            const size_t qrow = (size_t)b * L + (size_t)qb * BQ + wv * 16 + v + 8 * hf;
            __builtin_nontemporal_store(oacc[c][v], &Og[qrow * D + c * 16 + ln]);
        }
    }
}

extern "C" void kernel_launch(void* const* d_in, const int* in_sizes, int n_in,
                              void* d_out, int out_size, void* d_ws, size_t ws_size,
                              hipStream_t stream) {
    (void)in_sizes; (void)n_in; (void)out_size; (void)d_ws; (void)ws_size;
    constexpr int B = 16, L = 2048, D = 64, BQ = 128;

    const float* q = (const float*)d_in[0];
    const float* k = (const float*)d_in[1];
    const float* v = (const float*)d_in[2];
    float* out  = (float*)d_out;                 // [B,L,D] output
    float* attn = out + (size_t)B * L * D;       // [B,L,L] attention

    dim3 grid(B * (L / BQ));                     // 256 blocks
    dim3 block(256);                             // 8 wave32 waves
    hipLaunchKernelGGL(ScaledDotProductAttention_90804198572610_kernel,
                       grid, block, 0, stream, q, k, v, out, attn);
}
